// MSAOverflowBuffer_45595372814975
// MI455X (gfx1250) — matrix operand order repaired and checked by reading, not compiled, not run on hardware
//
#include <hip/hip_runtime.h>
#include <math.h>

// ---------------- problem constants (match reference) ----------------
#define D_DIM   1024
#define P_DIM   131072
#define B_DIM   4096
#define H_NUM   4
#define HD_DIM  256
#define TOPK    16
#define CHUNK   64
#define NC      (P_DIM / CHUNK)      // 2048 compressed prototypes
#define INV_HTEMP 2.5f               // 1/(H*TEMP) = 1/(4*0.1)

typedef __bf16 bf16;
typedef __attribute__((ext_vector_type(16))) __bf16 v16bf;
typedef __attribute__((ext_vector_type(8)))  __bf16 v8bf;
typedef __attribute__((ext_vector_type(8)))  float  v8f;

// ---------------- workspace layout (bytes), ~74 MB total ----------------
#define OFF_HBF   ((size_t)0)                         // 4096x1024 bf16   = 8 MB
#define OFF_WQR   (OFF_HBF  + (size_t)B_DIM*D_DIM*2)  // 1024x1024 bf16   = 2 MB
#define OFF_WKR   (OFF_WQR  + (size_t)D_DIM*D_DIM*2)
#define OFF_WOUT  (OFF_WKR  + (size_t)D_DIM*D_DIM*2)
#define OFF_PROTF (OFF_WOUT + (size_t)D_DIM*D_DIM*2)  // 2048x1024 f32    = 8 MB
#define OFF_PROTB (OFF_PROTF+ (size_t)NC*D_DIM*4)     // 2048x1024 bf16   = 4 MB
#define OFF_QRN   (OFF_PROTB+ (size_t)NC*D_DIM*2)     // 4096x1024 bf16   = 8 MB (reused for retrieved_bf)
#define OFF_KRN   (OFF_QRN  + (size_t)B_DIM*D_DIM*2)  // 2048x1024 bf16   = 4 MB
#define OFF_BIG   (OFF_KRN  + (size_t)NC*D_DIM*2)     // 32 MB: qr_f32 -> kr_f32 -> scores
#define OFF_TOPW  (OFF_BIG  + (size_t)B_DIM*NC*4)     // 4096x16 f32
#define OFF_TOPI  (OFF_TOPW + (size_t)B_DIM*TOPK*4)   // 4096x16 i32

// ---------------- f32 -> bf16 convert (4 elems/thread) ----------------
__global__ void cvt_f32_bf16_kernel(const float* __restrict__ in,
                                    bf16* __restrict__ out, int n4) {
  int i = (blockIdx.x * blockDim.x + threadIdx.x);
  if (i < n4) {
    float4 v = *(const float4*)(in + (size_t)i * 4);
    bf16* o = out + (size_t)i * 4;
    o[0] = (bf16)v.x; o[1] = (bf16)v.y; o[2] = (bf16)v.z; o[3] = (bf16)v.w;
  }
}

// ---------------- chunk compression + l2norm (bandwidth-bound) ----------------
// one block (256 thr) per chunk of 64 prototype rows; each thread owns 4 cols
__global__ void __launch_bounds__(256)
compress_kernel(const float* __restrict__ protos, const int* __restrict__ evid,
                float* __restrict__ outF, bf16* __restrict__ outB) {
  __shared__ float w[CHUNK];
  __shared__ float red[256];
  const int c = blockIdx.x;
  const int t = threadIdx.x;

  float ev = 0.f;
  if (t < CHUNK) ev = (float)evid[c * CHUNK + t] + 1e-8f;
  red[t] = (t < CHUNK) ? ev : 0.f;
  __syncthreads();
  for (int s = 128; s > 0; s >>= 1) { if (t < s) red[t] += red[t + s]; __syncthreads(); }
  const float tot = red[0];
  __syncthreads();
  if (t < CHUNK) w[t] = ev / tot;
  __syncthreads();

  const float* base = protos + (size_t)c * CHUNK * D_DIM + t * 4;
  float ax = 0.f, ay = 0.f, az = 0.f, aw = 0.f;
  #pragma unroll 4
  for (int r = 0; r < CHUNK; ++r) {
    float4 v = *(const float4*)(base + (size_t)r * D_DIM);
    const float wr = w[r];
    ax += wr * v.x; ay += wr * v.y; az += wr * v.z; aw += wr * v.w;
  }
  red[t] = ax * ax + ay * ay + az * az + aw * aw;
  __syncthreads();
  for (int s = 128; s > 0; s >>= 1) { if (t < s) red[t] += red[t + s]; __syncthreads(); }
  const float inv = 1.0f / fmaxf(sqrtf(red[0]), 1e-12f);

  float4 o = { ax * inv, ay * inv, az * inv, aw * inv };
  *(float4*)(outF + (size_t)c * D_DIM + t * 4) = o;
  bf16* ob = outB + (size_t)c * D_DIM + t * 4;
  ob[0] = (bf16)o.x; ob[1] = (bf16)o.y; ob[2] = (bf16)o.z; ob[3] = (bf16)o.w;
}

// ---------------- WMMA bf16 GEMM:  C[M,N] = A[M,K] * B[N,K]^T ----------------
// A fragment: 16-bit 16x32 layout — lane(0..15)=row M, VGPR-halves hold K
//   lane L<16 : K = {k0..k0+7, k0+16..k0+23};  lane L+16: K = {k0+8..+15, k0+24..+31}
__device__ inline v16bf load_a_frag(const bf16* __restrict__ A, int lda,
                                    int row, int k0, int lane) {
  const int r  = row + (lane & 15);
  const int kh = (lane & 16) ? 8 : 0;
  const bf16* p = A + (size_t)r * lda + (k0 + kh);
  v8bf lo = *(const v8bf*)p;
  v8bf hi = *(const v8bf*)(p + 16);
  return __builtin_shufflevector(lo, hi, 0, 1, 2, 3, 4, 5, 6, 7,
                                 8, 9, 10, 11, 12, 13, 14, 15);
}

// LDS B slab: 64 cols x 32 k of bf16, row stride 80 B (20 banks -> the 16
// fragment lanes hit disjoint 4-bank groups). Double buffered.
#define BSTRIDE 80
#define BSLAB   (64 * BSTRIDE)   // 5120 B per buffer

// B fragment (32x16, K x N): lane(0..15)=col N holds K = {0..7, 16..23}, lane+16 holds {8..15, 24..31}
__device__ inline v16bf lds_b_frag(const unsigned char* sbuf, int nt, int lane) {
  const int c  = nt * 16 + (lane & 15);
  const int kh = (lane & 16) ? 16 : 0;              // bytes (8 halves)
  const unsigned char* p = sbuf + c * BSTRIDE + kh; // 16B aligned
  v8bf lo = *(const v8bf*)p;
  v8bf hi = *(const v8bf*)(p + 32);
  return __builtin_shufflevector(lo, hi, 0, 1, 2, 3, 4, 5, 6, 7,
                                 8, 9, 10, 11, 12, 13, 14, 15);
}

// async copy 16 B global -> LDS (tracked by ASYNCcnt)
__device__ inline void async_g2l_b128(unsigned lds_off, const bf16* g) {
  asm volatile("global_load_async_to_lds_b128 %0, %1, off"
               :: "v"(lds_off), "v"(g) : "memory");
}
__device__ inline void wait_asynccnt0() {
  asm volatile("s_wait_asynccnt 0" ::: "memory");
}

// block = 256 thr = 8 waves; block tile 128(M) x 64(N); wave tile 16 x 64.
// B slab staged in LDS via double-buffered async global->LDS copies.
__global__ void __launch_bounds__(256)
gemm_bf16_nt_kernel(const bf16* __restrict__ A, const bf16* __restrict__ Bm,
                    float* __restrict__ C, int M, int N, int K) {
  __shared__ __align__(16) unsigned char sB[2][BSLAB];

  const int t    = threadIdx.x;
  const int lane = t & 31;
  const int wave = t >> 5;
  const int row  = blockIdx.y * 128 + wave * 16;
  const int col0 = blockIdx.x * 64;

  // staging assignment: thread t copies 16 B: slab row (t>>2), segment (t&3)
  const int srow = t >> 2;      // 0..63 (column of B tile)
  const int sseg = t & 3;       // 4 x 16B segments = 32 halves
  const bf16* gB = Bm + (size_t)(col0 + srow) * K + sseg * 8;
  const unsigned ldsRow = (unsigned)(srow * BSTRIDE + sseg * 16);
  const unsigned ldsBase0 = (unsigned)(size_t)&sB[0][0];

  const int nk = K >> 5;
  // prefetch slab 0
  async_g2l_b128(ldsBase0 + ldsRow, gB);

  v8f acc0 = {}, acc1 = {}, acc2 = {}, acc3 = {};
  for (int i = 0; i < nk; ++i) {
    wait_asynccnt0();       // my writes of slab i complete
    __syncthreads();        // all waves' slab-i writes visible; prev buf free
    if (i + 1 < nk)         // prefetch slab i+1 into other buffer
      async_g2l_b128(ldsBase0 + (unsigned)(((i + 1) & 1) * BSLAB) + ldsRow,
                     gB + (size_t)(i + 1) * 32);

    const unsigned char* sbuf = &sB[i & 1][0];
    v16bf a  = load_a_frag(A, K, row, i * 32, lane);
    v16bf b0 = lds_b_frag(sbuf, 0, lane);
    v16bf b1 = lds_b_frag(sbuf, 1, lane);
    v16bf b2 = lds_b_frag(sbuf, 2, lane);
    v16bf b3 = lds_b_frag(sbuf, 3, lane);
    acc0 = __builtin_amdgcn_wmma_f32_16x16x32_bf16(false, a, false, b0, (short)0, acc0, false, false);
    acc1 = __builtin_amdgcn_wmma_f32_16x16x32_bf16(false, a, false, b1, (short)0, acc1, false, false);
    acc2 = __builtin_amdgcn_wmma_f32_16x16x32_bf16(false, a, false, b2, (short)0, acc2, false, false);
    acc3 = __builtin_amdgcn_wmma_f32_16x16x32_bf16(false, a, false, b3, (short)0, acc3, false, false);
  }

  // C/D layout: VGPR v, lane<16 -> M=v, lane>=16 -> M=v+8; N = lane&15
  const int rofs = (lane & 16) ? 8 : 0;
  const int cl   = lane & 15;
  v8f accs[4] = { acc0, acc1, acc2, acc3 };
  #pragma unroll
  for (int nt = 0; nt < 4; ++nt) {
    float* cp = C + (size_t)(row + rofs) * N + col0 + nt * 16 + cl;
    #pragma unroll
    for (int v = 0; v < 8; ++v) cp[(size_t)v * N] = accs[nt][v];
  }
}

// ---------------- per-head l2norm (+optional scale), f32 -> bf16 ----------------
// one block per row; 4 heads x 256 cols; thread t owns cols 4t..4t+3 (head = t>>6)
__global__ void __launch_bounds__(256)
headnorm_kernel(const float* __restrict__ X, bf16* __restrict__ Y, float scale) {
  __shared__ float red[256];
  const int rowi = blockIdx.x;
  const int t = threadIdx.x;
  float4 v = *(const float4*)(X + (size_t)rowi * D_DIM + t * 4);
  red[t] = v.x * v.x + v.y * v.y + v.z * v.z + v.w * v.w;
  __syncthreads();
  const int lt = t & 63;
  for (int s = 32; s > 0; s >>= 1) { if (lt < s) red[t] += red[t + s]; __syncthreads(); }
  const float n   = sqrtf(red[(t >> 6) << 6]);
  const float inv = scale / fmaxf(n, 1e-12f);
  bf16* yp = Y + (size_t)rowi * D_DIM + t * 4;
  yp[0] = (bf16)(v.x * inv); yp[1] = (bf16)(v.y * inv);
  yp[2] = (bf16)(v.z * inv); yp[3] = (bf16)(v.w * inv);
}

// ---------------- top-16 (iterative argmax in LDS) + softmax ----------------
__global__ void __launch_bounds__(256)
topk_softmax_kernel(const float* __restrict__ S, float* __restrict__ W,
                    int* __restrict__ I) {
  __shared__ float sv[NC];
  __shared__ float rv[256];
  __shared__ int   ri[256];
  __shared__ float topv[TOPK];
  __shared__ int   topi[TOPK];
  const int row = blockIdx.x, t = threadIdx.x;
  const float* sr = S + (size_t)row * NC;
  for (int i = t; i < NC; i += 256) sv[i] = sr[i];
  __syncthreads();
  for (int k = 0; k < TOPK; ++k) {
    float bv = -3.4e38f; int bi = 0;
    for (int i = t; i < NC; i += 256) { float x = sv[i]; if (x > bv) { bv = x; bi = i; } }
    rv[t] = bv; ri[t] = bi;
    __syncthreads();
    for (int s = 128; s > 0; s >>= 1) {
      if (t < s && rv[t + s] > rv[t]) { rv[t] = rv[t + s]; ri[t] = ri[t + s]; }
      __syncthreads();
    }
    if (t == 0) { topv[k] = rv[0]; topi[k] = ri[0]; sv[ri[0]] = -3.4e38f; }
    __syncthreads();
  }
  if (t == 0) {
    const float m = topv[0];
    float e[TOPK]; float sum = 0.f;
    for (int k = 0; k < TOPK; ++k) { e[k] = __expf(topv[k] - m); sum += e[k]; }
    const float is = 1.f / sum;
    for (int k = 0; k < TOPK; ++k) {
      W[row * TOPK + k] = e[k] * is;
      I[row * TOPK + k] = topi[k];
    }
  }
}

// ---------------- gather top-k prototypes & blend -> bf16 ----------------
__global__ void __launch_bounds__(256)
blend_kernel(const float* __restrict__ protosF, const float* __restrict__ W,
             const int* __restrict__ I, bf16* __restrict__ R) {
  const int row = blockIdx.x, t = threadIdx.x;
  float ax = 0.f, ay = 0.f, az = 0.f, aw = 0.f;
  #pragma unroll
  for (int k = 0; k < TOPK; ++k) {
    const float wgt = W[row * TOPK + k];
    const int   idx = I[row * TOPK + k];
    float4 v = *(const float4*)(protosF + (size_t)idx * D_DIM + t * 4);
    ax += wgt * v.x; ay += wgt * v.y; az += wgt * v.z; aw += wgt * v.w;
  }
  bf16* rp = R + (size_t)row * D_DIM + t * 4;
  rp[0] = (bf16)ax; rp[1] = (bf16)ay; rp[2] = (bf16)az; rp[3] = (bf16)aw;
}

// ---------------- driver ----------------
extern "C" void kernel_launch(void* const* d_in, const int* in_sizes, int n_in,
                              void* d_out, int out_size, void* d_ws, size_t ws_size,
                              hipStream_t stream) {
  (void)in_sizes; (void)n_in; (void)out_size; (void)ws_size;
  const float* h      = (const float*)d_in[0];
  const float* protos = (const float*)d_in[1];
  const int*   evid   = (const int*)d_in[2];
  const float* W_QR   = (const float*)d_in[3];
  const float* W_KR   = (const float*)d_in[4];
  const float* W_out  = (const float*)d_in[5];
  float* out = (float*)d_out;
  char*  ws  = (char*)d_ws;

  bf16*  h_bf    = (bf16*)(ws + OFF_HBF);
  bf16*  wqr_bf  = (bf16*)(ws + OFF_WQR);
  bf16*  wkr_bf  = (bf16*)(ws + OFF_WKR);
  bf16*  wout_bf = (bf16*)(ws + OFF_WOUT);
  float* protF   = (float*)(ws + OFF_PROTF);
  bf16*  protB   = (bf16*)(ws + OFF_PROTB);
  bf16*  qrn_bf  = (bf16*)(ws + OFF_QRN);    // later reused as retrieved_bf
  bf16*  krn_bf  = (bf16*)(ws + OFF_KRN);
  float* big     = (float*)(ws + OFF_BIG);   // qr_f32 -> kr_f32 -> scores
  float* topw    = (float*)(ws + OFF_TOPW);
  int*   topi    = (int*)(ws + OFF_TOPI);

  // 1) bf16 conversions
  {
    int n4 = (B_DIM * D_DIM) / 4;
    cvt_f32_bf16_kernel<<<(n4 + 255) / 256, 256, 0, stream>>>(h, h_bf, n4);
    int w4 = (D_DIM * D_DIM) / 4;
    cvt_f32_bf16_kernel<<<(w4 + 255) / 256, 256, 0, stream>>>(W_QR, wqr_bf, w4);
    cvt_f32_bf16_kernel<<<(w4 + 255) / 256, 256, 0, stream>>>(W_KR, wkr_bf, w4);
    cvt_f32_bf16_kernel<<<(w4 + 255) / 256, 256, 0, stream>>>(W_out, wout_bf, w4);
  }
  // 2) chunk compression (dominant HBM traffic: 512 MB read)
  compress_kernel<<<NC, 256, 0, stream>>>(protos, evid, protF, protB);

  // 3) qr = h @ W_QR^T ; per-head l2norm * 1/(H*TEMP) -> bf16
  gemm_bf16_nt_kernel<<<dim3(D_DIM / 64, B_DIM / 128), 256, 0, stream>>>(
      h_bf, wqr_bf, big, B_DIM, D_DIM, D_DIM);
  headnorm_kernel<<<B_DIM, 256, 0, stream>>>(big, qrn_bf, INV_HTEMP);

  // 4) kr = protos' @ W_KR^T ; per-head l2norm -> bf16
  gemm_bf16_nt_kernel<<<dim3(D_DIM / 64, NC / 128), 256, 0, stream>>>(
      protB, wkr_bf, big, NC, D_DIM, D_DIM);
  headnorm_kernel<<<NC, 256, 0, stream>>>(big, krn_bf, 1.0f);

  // 5) scores[B, NC] = qrn @ krn^T   (scale already folded into qrn)
  gemm_bf16_nt_kernel<<<dim3(NC / 64, B_DIM / 128), 256, 0, stream>>>(
      qrn_bf, krn_bf, big, B_DIM, NC, D_DIM);

  // 6) top-16 + softmax
  topk_softmax_kernel<<<B_DIM, 256, 0, stream>>>(big, topw, topi);

  // 7) blend prototypes -> retrieved (bf16, reuses qrn buffer)
  blend_kernel<<<B_DIM, 256, 0, stream>>>(protF, topw, topi, qrn_bf);

  // 8) out = retrieved @ W_out^T
  gemm_bf16_nt_kernel<<<dim3(D_DIM / 64, B_DIM / 128), 256, 0, stream>>>(
      qrn_bf, wout_bf, out, B_DIM, D_DIM, D_DIM);
}
